// BahdanauAttention_11184094838827
// MI455X (gfx1250) — matrix operand order repaired and checked by reading, not compiled, not run on hardware
//
#include <hip/hip_runtime.h>
#include <hip/hip_bf16.h>
#include <math.h>

#define B_SZ   32
#define KLEN   4096
#define H      512
#define KT     128               // key rows per tile
#define NTILES (KLEN / KT)       // 32
#define NB     128               // N columns per staged Wk block
#define KROW   (H + 8)           // padded bf16 row stride for keys tile (bank-conflict-free)
#define WROW   (NB + 8)          // padded bf16 row stride for Wk block

typedef __attribute__((ext_vector_type(16))) __bf16 v16bf;
typedef __attribute__((ext_vector_type(8)))  __bf16 v8bf;
typedef __attribute__((ext_vector_type(4)))  __bf16 v4bf;
typedef __attribute__((ext_vector_type(8)))  float  v8f;

union ABFrag { v16bf v; v8bf h[2]; };

// LDS layout (dynamic shared memory, bytes)
constexpr int S_KEYS = 0;
constexpr int S_WKB  = S_KEYS + KT * KROW * 2;   // 133120
constexpr int S_QH   = S_WKB  + H  * WROW * 2;   // 272384
constexpr int S_WV   = S_QH   + H * 4;           // 274432
constexpr int S_S    = S_WV   + H * 4;           // 276480
constexpr int S_P    = S_S    + KT * 4;          // 276992
constexpr int S_RED  = S_P    + KT * 4;          // 277504
constexpr int SMEM_BYTES = S_RED + 64;           // 277568  (< 320KB WGP LDS)

// ---------------------------------------------------------------------------
// Kernel 0: qh[b,:] = query[b,:] @ Wq   (tiny: 32x512 @ 512x512)
// ---------------------------------------------------------------------------
__global__ void __launch_bounds__(512)
qproj_kernel(const float* __restrict__ query,
             const float* __restrict__ Wq,
             float* __restrict__ qh)
{
    const int b = blockIdx.x;
    const int h = threadIdx.x;
    const float* q = query + b * H;
    float acc = 0.f;
    for (int d = 0; d < H; ++d)
        acc += q[d] * Wq[(size_t)d * H + h];   // coalesced across h
    qh[b * H + h] = acc;
}

// ---------------------------------------------------------------------------
// Kernel 1: per (batch, key-tile) fused GEMM + tanh-score + tile softmax +
//           partial context. Keys read from HBM exactly once.
// ---------------------------------------------------------------------------
__global__ void __launch_bounds__(256)
attn_tile_kernel(const float* __restrict__ keys,
                 const float* __restrict__ Wk,
                 const float* __restrict__ qh,
                 const float* __restrict__ Wv,
                 float* __restrict__ pbuf,    // [B, KLEN] unnormalized exp(s - m_tile)
                 float* __restrict__ mlbuf,   // [B, NTILES] float2 (m_tile, l_tile)
                 float* __restrict__ ctxp)    // [B, NTILES, H] partial contexts
{
    extern __shared__ char smem[];
    __bf16* sKeys = (__bf16*)(smem + S_KEYS);   // KT x KROW bf16
    __bf16* sWkB  = (__bf16*)(smem + S_WKB);    // H  x WROW bf16 (one NB block)
    float*  sQh   = (float*)(smem + S_QH);
    float*  sWv   = (float*)(smem + S_WV);
    float*  sS    = (float*)(smem + S_S);
    float*  sP    = (float*)(smem + S_P);
    float*  sRed  = (float*)(smem + S_RED);

    const int tile = blockIdx.x;
    const int b    = blockIdx.y;
    const int tid  = threadIdx.x;
    const int lane = tid & 31;
    const int wid  = tid >> 5;

    const float* keysBase = keys + ((size_t)b * KLEN + (size_t)tile * KT) * H;

    // ---- stage keys tile (fp32 -> bf16) into LDS, float4 coalesced loads ----
    for (int i = tid; i < KT * (H / 4); i += 256) {
        int k  = i >> 7;           // / 128 float4s per row
        int c4 = i & 127;
        float4 f = ((const float4*)(keysBase + (size_t)k * H))[c4];
        v4bf o = { (__bf16)f.x, (__bf16)f.y, (__bf16)f.z, (__bf16)f.w };
        *(v4bf*)(sKeys + k * KROW + c4 * 4) = o;
    }
    // ---- stage qh[b] and Wv (fp32) ----
    if (tid < 128) {
        ((float4*)sQh)[tid] = ((const float4*)(qh + (size_t)b * H))[tid];
        ((float4*)sWv)[tid] = ((const float4*)Wv)[tid];
    }

    const int rbase = wid * 16;            // this wave's 16 output rows
    float rowAcc[8] = {0.f,0.f,0.f,0.f,0.f,0.f,0.f,0.f};

    for (int nbi = 0; nbi < H / NB; ++nbi) {
        const int nb = nbi * NB;
        __syncthreads();                   // prior-block LDS reads done (also covers keys staging)
        // ---- stage Wk[:, nb:nb+NB] (fp32 -> bf16), no transpose needed ----
        for (int i = tid; i < H * (NB / 4); i += 256) {
            int k  = i >> 5;               // / 32 float4s per row slice
            int c4 = i & 31;
            float4 f = ((const float4*)(Wk + (size_t)k * H + nb))[c4];
            v4bf o = { (__bf16)f.x, (__bf16)f.y, (__bf16)f.z, (__bf16)f.w };
            *(v4bf*)(sWkB + k * WROW + c4 * 4) = o;
        }
        __syncthreads();

        v8f acc[8];
        v8f zero = {};
        #pragma unroll
        for (int t = 0; t < 8; ++t) acc[t] = zero;

        // soft-prefetch the next Wk column block while this one computes
        if (nbi + 1 < H / NB)
            __builtin_prefetch(Wk + (size_t)(tid * 2) * H + nb + NB, 0, 0);

        const int half = lane >> 4;
        const int r    = lane & 15;
        for (int kc = 0; kc < H; kc += 32) {
            // A fragment: lane = M row; K 0-7/16-23 (lanes 0-15) or 8-15/24-31 (16-31)
            ABFrag a;
            const __bf16* ap = sKeys + (rbase + r) * KROW + kc + half * 8;
            a.h[0] = *(const v8bf*)(ap);
            a.h[1] = *(const v8bf*)(ap + 16);
            #pragma unroll
            for (int nt = 0; nt < 8; ++nt) {
                // B fragment: lane = N col; K 0-7/16-23 (lanes 0-15) or 8-15/24-31 (16-31)
                ABFrag bb;
                const __bf16* bp = sWkB + (kc + half * 8) * WROW + nt * 16 + r;
                #pragma unroll
                for (int s = 0; s < 8; ++s) {
                    bb.v[s]     = bp[s * WROW];
                    bb.v[8 + s] = bp[(16 + s) * WROW];
                }
                acc[nt] = __builtin_amdgcn_wmma_f32_16x16x32_bf16(
                    false, a.v, false, bb.v, (short)0, acc[nt], false, false);
            }
        }

        // ---- epilogue: s_row += sum_n tanh(qh[n] + C[row,n]) * Wv[n] ----
        const int nl = lane & 15;
        #pragma unroll
        for (int nt = 0; nt < 8; ++nt) {
            const int n = nb + nt * 16 + nl;
            const float qn = sQh[n];
            const float wn = sWv[n];
            #pragma unroll
            for (int v = 0; v < 8; ++v)
                rowAcc[v] += tanhf(qn + acc[nt][v]) * wn;
        }
    }

    // reduce over the 16 N-lanes of each half-wave (rows differ by lane half)
    #pragma unroll
    for (int v = 0; v < 8; ++v) {
        float x = rowAcc[v];
        x += __shfl_xor(x, 1, 32);
        x += __shfl_xor(x, 2, 32);
        x += __shfl_xor(x, 4, 32);
        x += __shfl_xor(x, 8, 32);
        rowAcc[v] = x;
    }
    if ((lane & 15) == 0) {
        #pragma unroll
        for (int v = 0; v < 8; ++v)
            sS[rbase + (lane >> 4) * 8 + v] = rowAcc[v];
    }
    __syncthreads();

    // ---- tile-local softmax partials: m = max, p = exp(s - m), l = sum p ----
    float sv = (tid < KT) ? sS[tid] : -3.3e38f;
    float mx = sv;
    mx = fmaxf(mx, __shfl_xor(mx, 1, 32));
    mx = fmaxf(mx, __shfl_xor(mx, 2, 32));
    mx = fmaxf(mx, __shfl_xor(mx, 4, 32));
    mx = fmaxf(mx, __shfl_xor(mx, 8, 32));
    mx = fmaxf(mx, __shfl_xor(mx, 16, 32));
    if (lane == 0) sRed[wid] = mx;
    __syncthreads();
    if (tid == 0) {
        float m = sRed[0];
        for (int w = 1; w < 8; ++w) m = fmaxf(m, sRed[w]);
        sRed[8] = m;
    }
    __syncthreads();
    const float M = sRed[8];
    float p = 0.f;
    if (tid < KT) {
        p = __expf(sS[tid] - M);
        sP[tid] = p;
        pbuf[(size_t)b * KLEN + (size_t)tile * KT + tid] = p;
    }
    float ps = p;
    ps += __shfl_xor(ps, 1, 32);
    ps += __shfl_xor(ps, 2, 32);
    ps += __shfl_xor(ps, 4, 32);
    ps += __shfl_xor(ps, 8, 32);
    ps += __shfl_xor(ps, 16, 32);
    if (lane == 0) sRed[wid] = ps;
    __syncthreads();
    if (tid == 0) {
        float L = 0.f;
        for (int w = 0; w < 8; ++w) L += sRed[w];
        ((float2*)mlbuf)[(size_t)b * NTILES + tile] = make_float2(M, L);
    }
    __syncthreads();

    // ---- partial context from the resident LDS keys tile (keys read once) ----
    // all 32 lanes read the same k, consecutive h -> contiguous, conflict-free
    const int h0 = tid * 2;
    float a0 = 0.f, a1 = 0.f;
    for (int k = 0; k < KT; ++k) {
        const float pk = sP[k];
        const __bf16* kp = sKeys + k * KROW + h0;
        a0 += pk * (float)kp[0];
        a1 += pk * (float)kp[1];
    }
    float* cp = ctxp + ((size_t)b * NTILES + tile) * H;
    cp[h0]     = a0;
    cp[h0 + 1] = a1;
}

// ---------------------------------------------------------------------------
// Kernel 2: combine tiles -> global softmax; write weights then context
// ---------------------------------------------------------------------------
__global__ void __launch_bounds__(256)
combine_kernel(const float* __restrict__ pbuf,
               const float* __restrict__ mlbuf,
               const float* __restrict__ ctxp,
               float* __restrict__ out)
{
    __shared__ float sScale[NTILES];
    const int b   = blockIdx.x;
    const int tid = threadIdx.x;

    if (tid < 32) {                 // wave 0: one lane per tile
        float2 ml = ((const float2*)mlbuf)[(size_t)b * NTILES + tid];
        float m = ml.x, l = ml.y;
        float M = m;
        M = fmaxf(M, __shfl_xor(M, 1, 32));
        M = fmaxf(M, __shfl_xor(M, 2, 32));
        M = fmaxf(M, __shfl_xor(M, 4, 32));
        M = fmaxf(M, __shfl_xor(M, 8, 32));
        M = fmaxf(M, __shfl_xor(M, 16, 32));
        float e = __expf(m - M);
        float L = l * e;
        L += __shfl_xor(L, 1, 32);
        L += __shfl_xor(L, 2, 32);
        L += __shfl_xor(L, 4, 32);
        L += __shfl_xor(L, 8, 32);
        L += __shfl_xor(L, 16, 32);
        sScale[tid] = e / L;
    }
    __syncthreads();

    // weights = p * exp(m_t - M) / L   -> out[B*H : B*H + B*KLEN]
    const float* pb = pbuf + (size_t)b * KLEN;
    float* wout = out + (size_t)B_SZ * H + (size_t)b * KLEN;
    for (int i = tid; i < KLEN; i += 256)
        wout[i] = pb[i] * sScale[i >> 7];          // KT = 128 keys per tile

    // context = sum_t ctxp_t * exp(m_t - M) / L  -> out[0 : B*H]
    float* cout = out + (size_t)b * H;
    for (int h = tid; h < H; h += 256) {
        float c = 0.f;
        for (int t = 0; t < NTILES; ++t)
            c += ctxp[((size_t)b * NTILES + t) * H + h] * sScale[t];
        cout[h] = c;
    }
}

// ---------------------------------------------------------------------------
extern "C" void kernel_launch(void* const* d_in, const int* in_sizes, int n_in,
                              void* d_out, int out_size, void* d_ws, size_t ws_size,
                              hipStream_t stream) {
    const float* query = (const float*)d_in[0];   // (32, 1, 512)
    const float* keys  = (const float*)d_in[1];   // (32, 4096, 512)
    const float* Wq    = (const float*)d_in[2];   // (512, 512)
    const float* Wk    = (const float*)d_in[3];   // (512, 512)
    const float* Wv    = (const float*)d_in[4];   // (512, 1)
    float* out = (float*)d_out;                   // context (32*512) ++ weights (32*4096)
    float* ws  = (float*)d_ws;

    float* qh    = ws;                              // 32*512
    float* pbuf  = qh    + B_SZ * H;                // 32*4096
    float* mlbuf = pbuf  + B_SZ * KLEN;             // 32*32*2 (float2 pairs)
    float* ctxp  = mlbuf + B_SZ * NTILES * 2;       // 32*32*512
    // total workspace: ~2.6 MB of floats

    qproj_kernel<<<B_SZ, H, 0, stream>>>(query, Wq, qh);

    dim3 grid(NTILES, B_SZ);                        // 32 x 32 workgroups
    attn_tile_kernel<<<grid, 256, SMEM_BYTES, stream>>>(keys, Wk, qh, Wv,
                                                        pbuf, mlbuf, ctxp);

    combine_kernel<<<B_SZ, 256, 0, stream>>>(pbuf, mlbuf, ctxp, out);
}
